// SimpleCINN_22419729285719
// MI455X (gfx1250) — matrix operand order repaired and checked by reading, not compiled, not run on hardware
//
#include <hip/hip_runtime.h>

// CDNA5 / gfx1250, wave32. WMMA-based coupling-flow forward.
// Per wave: 16 points (M=16). Layer1 uses f32 WMMA 16x16x4 (exact).
// Layers 2/3 use split-f16 WMMA 16x16x32 with f32 accum (3-term Markidis).
// Activations live in LDS as f32; the split-f16 A-fragment build uses packed
// v_cvt_pk_rtz conversions. Weight B-fragments are pre-swizzled into d_ws.

typedef __attribute__((ext_vector_type(16))) _Float16 v16h;
typedef __attribute__((ext_vector_type(2)))  __fp16   v2g;   // pkrtz result type
typedef __attribute__((ext_vector_type(8)))  float    v8f;
typedef __attribute__((ext_vector_type(2)))  float    v2f;

#define N_PTS   1000000
#define NB      8
#define NTILES  (N_PTS / 16)   // 62500 exactly

// d_ws layout (bytes):
//  [0      ,   8192) L1 f32 B-frags  [8][4][32][2]            f32
//  [8192   , 139264) L2 f16 B-frags  [8][2][4][2][32][16]     half (hi,lo)
//  [139264 , 147456) L3 f16 B-frags  [8][2][2][32][16]        half (hi,lo)
//  [147456 , 147584) scale [8][4] f32
//  [147584 , 147616) lsum  [8]    f32
#define WS_L1   0
#define WS_L2   8192
#define WS_L3   139264
#define WS_SCL  147456
#define WS_LSUM 147584

// Intra-wave LDS RAW needs no hardware wait (DS ops complete in order per
// wave); we only need to stop the compiler from reordering across phases.
__device__ __forceinline__ void lds_cbar() {
    asm volatile("" ::: "memory");
}

__device__ __forceinline__ v8f splat8(float v) {
    v8f r;
#pragma unroll
    for (int k = 0; k < 8; ++k) r[k] = v;
    return r;
}

// Build split-f16 A fragments (16x32 f16 layout) from one f32 row of hbuf.
// Lane l: row m=l%16; kb = kt*32 + (l<16?0:8); halves j: k=kb+j (j<8),
// k=kb+16+(j-8) (j>=8). Packed pkrtz converts value-pairs straight into the
// half2 dwords of the fragment.
__device__ __forceinline__ void split16(const float* __restrict__ hr, int kb,
                                        v16h& ahi, v16h& alo) {
    union { v16h v; v2g p[8]; } A, L;
#pragma unroll
    for (int t = 0; t < 4; ++t) {
        {
            float w0 = hr[kb + 2 * t], w1 = hr[kb + 2 * t + 1];
            v2g h = __builtin_amdgcn_cvt_pkrtz(w0, w1);
            A.p[t] = h;
            L.p[t] = __builtin_amdgcn_cvt_pkrtz(w0 - (float)h[0], w1 - (float)h[1]);
        }
        {
            float w0 = hr[kb + 16 + 2 * t], w1 = hr[kb + 16 + 2 * t + 1];
            v2g h = __builtin_amdgcn_cvt_pkrtz(w0, w1);
            A.p[4 + t] = h;
            L.p[4 + t] = __builtin_amdgcn_cvt_pkrtz(w0 - (float)h[0], w1 - (float)h[1]);
        }
    }
    ahi = A.v; alo = L.v;
}

// 3-term split-f16 accumulate: acc += Ahi*Bhi + Alo*Bhi + Ahi*Blo
__device__ __forceinline__ v8f wmma3(v8f acc, v16h ahi, v16h alo,
                                     const _Float16* __restrict__ bp) {
    v16h bhi = *reinterpret_cast<const v16h*>(bp);
    v16h blo = *reinterpret_cast<const v16h*>(bp + 512);
    acc = __builtin_amdgcn_wmma_f32_16x16x32_f16(false, ahi, false, bhi, (short)0, acc, false, false);
    acc = __builtin_amdgcn_wmma_f32_16x16x32_f16(false, alo, false, bhi, (short)0, acc, false, false);
    acc = __builtin_amdgcn_wmma_f32_16x16x32_f16(false, ahi, false, blo, (short)0, acc, false, false);
    return acc;
}

// ---------------- weight prep: build WMMA B-fragments ----------------
__global__ void prep_kernel(const float* __restrict__ W1, const float* __restrict__ W2,
                            const float* __restrict__ W3, const float* __restrict__ gs,
                            float* __restrict__ l1f, _Float16* __restrict__ l2f,
                            _Float16* __restrict__ l3f, float* __restrict__ scl,
                            float* __restrict__ lsum) {
    int t = blockIdx.x * blockDim.x + threadIdx.x;
    // Layer2: B[k][n] = W2[i][n][k]; lane l: n=l%16, k = kt*32 + (l<16?0:8) + (j<8 ? j : 16+(j-8)).
    if (t < 32768) {
        int j = t & 15, lane = (t >> 4) & 31, nt = (t >> 9) & 3, kt = (t >> 11) & 1, i = t >> 12;
        int n = nt * 16 + (lane & 15);
        int k = kt * 32 + ((lane >> 4) ? 8 : 0) + (j & 7) + ((j >> 3) << 4);
        float w = W2[(i * 64 + n) * 64 + k];
        _Float16 hi = (_Float16)w;
        _Float16 lo = (_Float16)(w - (float)hi);
        int bidx = ((i * 2 + kt) * 4 + nt) * 2;
        l2f[((bidx + 0) * 32 + lane) * 16 + j] = hi;
        l2f[((bidx + 1) * 32 + lane) * 16 + j] = lo;
    }
    // Layer1: f32 B 4x16, 2 VGPRs: lane l: n=l%16, k = (l<16?0:2)+j.
    if (t < 2048) {
        int j = t & 1, lane = (t >> 1) & 31, nt = (t >> 6) & 3, i = t >> 8;
        int n = nt * 16 + (lane & 15);
        int k = ((lane >> 4) ? 2 : 0) + j;
        l1f[((i * 4 + nt) * 32 + lane) * 2 + j] = W1[(i * 64 + n) * 4 + k];
    }
    // Layer3: B[k][n] = W3[i][n][k], n padded 4->16 with zeros.
    if (t < 8192) {
        int j = t & 15, lane = (t >> 4) & 31, kt = (t >> 9) & 1, i = t >> 10;
        int n = lane & 15;
        int k = kt * 32 + ((lane >> 4) ? 8 : 0) + (j & 7) + ((j >> 3) << 4);
        float w = (n < 4) ? W3[(i * 4 + n) * 64 + k] : 0.f;
        _Float16 hi = (_Float16)w;
        _Float16 lo = (_Float16)(w - (float)hi);
        int bidx = (i * 2 + kt) * 2;
        l3f[((bidx + 0) * 32 + lane) * 16 + j] = hi;
        l3f[((bidx + 1) * 32 + lane) * 16 + j] = lo;
    }
    if (t < 32) {
        int i = t >> 2, d = t & 3;
        scl[t] = 0.2f * log1pf(expf(0.5f * gs[i * 4 + d]));
    }
    if (t < 8) {
        float s = 0.f;
        for (int d = 0; d < 4; ++d)
            s += logf(0.2f * log1pf(expf(0.5f * gs[t * 4 + d])));
        lsum[t] = s;
    }
}

// ---------------- main flow kernel ----------------
__global__ void __launch_bounds__(256)
flow_kernel(const float* __restrict__ q, const float* __restrict__ Hc,
            const float* __restrict__ b1, const float* __restrict__ b2,
            const float* __restrict__ b3, const float* __restrict__ wperm,
            const float* __restrict__ goff,
            const float* __restrict__ l1f, const _Float16* __restrict__ l2f,
            const _Float16* __restrict__ l3f, const float* __restrict__ scl,
            const float* __restrict__ lsum, float* __restrict__ out) {
    __shared__ float hbuf[8][16][68];   // per-wave 16x64 activation tile (+pad)

    const int lane  = threadIdx.x & 31;
    const int wv    = threadIdx.x >> 5;
    const int gwave = (blockIdx.x * blockDim.x + threadIdx.x) >> 5;
    const int nwav  = (gridDim.x * blockDim.x) >> 5;
    const int hi16  = lane >> 4;        // 0 for lanes 0-15, 1 for 16-31
    const int lm    = lane & 15;
    const int m0    = hi16 * 8;
    float (*hb)[68] = hbuf[wv];

#pragma unroll 1
    for (int tile = gwave; tile < NTILES; tile += nwav) {
        const int p = tile * 16 + lm;
        float x0 = 0.f, x1 = 0.f, x2 = 0.f, x3 = 0.f;   // point state (lanes 0-15)
        float h0 = 0.f, h1 = 0.f;                        // conditioning (lanes 16-31)
        float logdet = 0.f;
        if (!hi16) {
            float4 xv = reinterpret_cast<const float4*>(q)[p];
            x0 = xv.x; x1 = xv.y; x2 = xv.z; x3 = xv.w;
        } else {
            float2 hv = reinterpret_cast<const float2*>(Hc)[p];
            h0 = hv.x; h1 = hv.y;
        }

#pragma unroll 1
        for (int i = 0; i < NB; ++i) {
            // ---- layer 1: [16x4] @ [4x64] via f32 WMMA 16x16x4 ----
            v2f A1;
            A1[0] = hi16 ? h0 : x0;
            A1[1] = hi16 ? h1 : x1;
            lds_cbar();
#pragma unroll
            for (int nt = 0; nt < 4; ++nt) {
                v2f B1 = *reinterpret_cast<const v2f*>(&l1f[((i * 4 + nt) * 32 + lane) * 2]);
                v8f acc = splat8(b1[i * 64 + nt * 16 + lm]);
                acc = __builtin_amdgcn_wmma_f32_16x16x4_f32(
                    false, A1, false, B1, (short)0, acc, false, false);
#pragma unroll
                for (int v = 0; v < 8; ++v)
                    hb[m0 + v][nt * 16 + lm] = fmaxf(acc[v], 0.f);   // relu -> h1
            }
            lds_cbar();

            // ---- layer 2: [16x64] @ [64x64], split-f16 WMMA ----
            {
                v8f acc0 = splat8(b2[i * 64 +      lm]);
                v8f acc1 = splat8(b2[i * 64 + 16 + lm]);
                v8f acc2 = splat8(b2[i * 64 + 32 + lm]);
                v8f acc3 = splat8(b2[i * 64 + 48 + lm]);
#pragma unroll
                for (int kt = 0; kt < 2; ++kt) {
                    v16h ahi, alo;
                    split16(hb[lm], kt * 32 + m0, ahi, alo);
                    const _Float16* bp = l2f + ((i * 2 + kt) * 8) * 512 + lane * 16;
                    acc0 = wmma3(acc0, ahi, alo, bp);
                    acc1 = wmma3(acc1, ahi, alo, bp + 1024);
                    acc2 = wmma3(acc2, ahi, alo, bp + 2048);
                    acc3 = wmma3(acc3, ahi, alo, bp + 3072);
                }
                lds_cbar();
#pragma unroll
                for (int v = 0; v < 8; ++v) {
                    hb[m0 + v][     lm] = fmaxf(acc0[v], 0.f);       // relu -> h2
                    hb[m0 + v][16 + lm] = fmaxf(acc1[v], 0.f);
                    hb[m0 + v][32 + lm] = fmaxf(acc2[v], 0.f);
                    hb[m0 + v][48 + lm] = fmaxf(acc3[v], 0.f);
                }
            }
            lds_cbar();

            // ---- layer 3: [16x64] @ [64x(4->16)], split-f16 WMMA ----
            {
                v8f acc = splat8((lm < 4) ? b3[i * 4 + lm] : 0.f);
#pragma unroll
                for (int kt = 0; kt < 2; ++kt) {
                    v16h ahi, alo;
                    split16(hb[lm], kt * 32 + m0, ahi, alo);
                    const _Float16* bp = l3f + ((i * 2 + kt) * 2) * 512 + lane * 16;
                    acc = wmma3(acc, ahi, alo, bp);
                }
                lds_cbar();
                if (lm < 4) {   // D: lane holds a[m=v+8*hi16][n=lm]
#pragma unroll
                    for (int v = 0; v < 8; ++v)
                        hb[m0 + v][lm] = acc[v] * 0.1f;   // a = (h@W3.T + b3)*0.1
                }
            }
            lds_cbar();

            // ---- per-point tail (lanes 0-15 own point lm) ----
            const float sc0 = scl[i * 4 + 0], sc1 = scl[i * 4 + 1];
            const float sc2 = scl[i * 4 + 2], sc3 = scl[i * 4 + 3];
            const float o0 = goff[i * 4 + 0], o1 = goff[i * 4 + 1];
            const float o2 = goff[i * 4 + 2], o3 = goff[i * 4 + 3];
            const float* P = &wperm[i * 16];   // w_perm[i][d][k]
            if (!hi16) {
                float a0 = hb[lm][0], a1 = hb[lm][1], a2 = hb[lm][2], a3 = hb[lm][3];
                float s0 = 2.f * tanhf(a0), s1 = 2.f * tanhf(a1);
                x2 = x2 * expf(s0) + a2;
                x3 = x3 * expf(s1) + a3;
                logdet += s0 + s1 + lsum[i];
                float z0 = x0 * sc0 + o0, z1 = x1 * sc1 + o1;
                float z2 = x2 * sc2 + o2, z3 = x3 * sc3 + o3;
                // y = z @ P.T  ->  y[d] = sum_k z[k] * P[d][k]
                x0 = z0 * P[0]  + z1 * P[1]  + z2 * P[2]  + z3 * P[3];
                x1 = z0 * P[4]  + z1 * P[5]  + z2 * P[6]  + z3 * P[7];
                x2 = z0 * P[8]  + z1 * P[9]  + z2 * P[10] + z3 * P[11];
                x3 = z0 * P[12] + z1 * P[13] + z2 * P[14] + z3 * P[15];
            }
            lds_cbar();
        }

        if (!hi16) {
            float4 o;
            o.x = x0; o.y = x1; o.z = x2; o.w = x3;
            reinterpret_cast<float4*>(out)[p] = o;
            out[4 * N_PTS + p] = logdet;
        }
    }
}

extern "C" void kernel_launch(void* const* d_in, const int* in_sizes, int n_in,
                              void* d_out, int out_size, void* d_ws, size_t ws_size,
                              hipStream_t stream) {
    (void)in_sizes; (void)n_in; (void)out_size; (void)ws_size;
    const float* q     = (const float*)d_in[0];
    const float* H     = (const float*)d_in[1];
    const float* W1    = (const float*)d_in[2];
    const float* b1    = (const float*)d_in[3];
    const float* W2    = (const float*)d_in[4];
    const float* b2    = (const float*)d_in[5];
    const float* W3    = (const float*)d_in[6];
    const float* b3    = (const float*)d_in[7];
    const float* wperm = (const float*)d_in[8];
    const float* gs    = (const float*)d_in[9];
    const float* go    = (const float*)d_in[10];

    char* ws = (char*)d_ws;
    float*    l1f  = (float*)(ws + WS_L1);
    _Float16* l2f  = (_Float16*)(ws + WS_L2);
    _Float16* l3f  = (_Float16*)(ws + WS_L3);
    float*    sclp = (float*)(ws + WS_SCL);
    float*    lsmp = (float*)(ws + WS_LSUM);

    prep_kernel<<<128, 256, 0, stream>>>(W1, W2, W3, gs, l1f, l2f, l3f, sclp, lsmp);
    flow_kernel<<<2048, 256, 0, stream>>>(q, H, b1, b2, b3, wperm, go,
                                          l1f, l2f, l3f, sclp, lsmp, (float*)d_out);
}
